// PerformerModel_44727789421273
// MI455X (gfx1250) — matrix operand order repaired and checked by reading, not compiled, not run on hardware
//
#include <hip/hip_runtime.h>
#include <hip/hip_bf16.h>
#include <stdint.h>

// Model constants
#define Bb 16
#define Ss 1024
#define Nt (Bb * Ss)      // 16384 tokens
#define DMm 512
#define Hh 8
#define Dd 64
#define Ff 256
#define Ll 4
#define LDS_STRIDE 40     // bf16 elems per LDS row (80B: 16B-aligned, skews banks)

typedef __attribute__((ext_vector_type(16))) __bf16 v16bf;
typedef __attribute__((ext_vector_type(8)))  float  v8f;

static __device__ __forceinline__ unsigned short f2bf(float x) {
  unsigned int u = __builtin_bit_cast(unsigned int, x);
  u += 0x7FFFu + ((u >> 16) & 1u);   // round-to-nearest-even
  return (unsigned short)(u >> 16);
}
static __device__ __forceinline__ float bf2f(unsigned short v) {
  unsigned int u = ((unsigned int)v) << 16;
  return __builtin_bit_cast(float, u);
}

union FragU { v16bf v; uint4 q[2]; };

// A-matrix 16x32 bf16 fragment from LDS tile laid out [m][k], row stride LDS_STRIDE.
// Lane l (0-15): M=l, K=0..7 (VGPR0-3) and K=16..23 (VGPR4-7).
// Lane l (16-31): M=l-16, K=8..15 and K=24..31.
static __device__ __forceinline__ v16bf frag_a(const unsigned short* lds, int mOff, int lane) {
  int m = lane & 15, hf = lane >> 4;
  const unsigned short* p = lds + (size_t)(mOff + m) * LDS_STRIDE + hf * 8;
  FragU f;
  f.q[0] = *(const uint4*)p;          // 16B aligned (row stride 80B, hf*16B)
  f.q[1] = *(const uint4*)(p + 16);
  return f.v;
}
// B-matrix 32x16 bf16 fragment from LDS tile laid out TRANSPOSED [n][k].
// Lane l (0-15): N=l, K=0..7 and K=16..23; lane l (16-31): N=l-16, K=8..15 and K=24..31.
static __device__ __forceinline__ v16bf frag_b(const unsigned short* lds, int nOff, int lane) {
  int n = lane & 15, hf = lane >> 4;
  const unsigned short* p = lds + (size_t)(nOff + n) * LDS_STRIDE + hf * 8;
  FragU f;
  f.q[0] = *(const uint4*)p;
  f.q[1] = *(const uint4*)(p + 16);
  return f.v;
}
static __device__ __forceinline__ v8f wmma_bf16(v16bf a, v16bf b, v8f c) {
  return __builtin_amdgcn_wmma_f32_16x16x32_bf16(false, a, false, b, (short)0, c, false, false);
}

// ---------------------------------------------------------------------------
// Generic GEMM: C[M,Nn] = act(A[M,K] @ W[K,Nn] + bias)
// f32 inputs, bf16 WMMA compute, f32 accumulate/output.
// Block tile 128x128, 8 waves of 64x32, K-step 32, double-buffered LDS.
// ACT: 0=none 1=relu 2=*scale (compile-time => branch-free epilogue)
// ---------------------------------------------------------------------------
template <int ACT, bool HAS_BIAS>
__global__ __launch_bounds__(256) void gemm_wmma(
    const float* __restrict__ A, const float* __restrict__ W,
    const float* __restrict__ bias, float* __restrict__ C,
    int M, int Nn, int K, float scale)
{
  __shared__ __align__(16) unsigned short lA[2][128 * LDS_STRIDE];
  __shared__ __align__(16) unsigned short lB[2][128 * LDS_STRIDE];
  const int tid = threadIdx.x;
  const int lane = tid & 31;
  const int wave = tid >> 5;
  const int mBase = blockIdx.y * 128;
  const int nBase = blockIdx.x * 128;
  const int mW = (wave >> 2) * 64;   // 2 wave rows
  const int nW = (wave & 3) * 32;    // 4 wave cols

  v8f acc[4][2] = {};

  auto load_tiles = [&](int kb, int buf) {
    // A tile 128x32 f32 -> bf16, LDS row-major [m][k]
#pragma unroll
    for (int p = 0; p < 4; ++p) {
      int idx = p * 256 + tid;
      int r = idx >> 3;
      int c4 = (idx & 7) * 4;
      float4 v = *(const float4*)(A + (size_t)(mBase + r) * K + kb + c4);
      unsigned int p0 = (unsigned int)f2bf(v.x) | ((unsigned int)f2bf(v.y) << 16);
      unsigned int p1 = (unsigned int)f2bf(v.z) | ((unsigned int)f2bf(v.w) << 16);
      *(uint2*)(&lA[buf][(size_t)r * LDS_STRIDE + c4]) = make_uint2(p0, p1);
    }
    // B tile 32x128 f32 -> bf16, transposed into LDS [n][k]
#pragma unroll
    for (int p = 0; p < 4; ++p) {
      int idx = p * 256 + tid;
      int k = idx >> 5;
      int n4 = (idx & 31) * 4;
      float4 v = *(const float4*)(W + (size_t)(kb + k) * Nn + nBase + n4);
      lB[buf][(size_t)(n4 + 0) * LDS_STRIDE + k] = f2bf(v.x);
      lB[buf][(size_t)(n4 + 1) * LDS_STRIDE + k] = f2bf(v.y);
      lB[buf][(size_t)(n4 + 2) * LDS_STRIDE + k] = f2bf(v.z);
      lB[buf][(size_t)(n4 + 3) * LDS_STRIDE + k] = f2bf(v.w);
    }
  };

  load_tiles(0, 0);
  __syncthreads();
  int cur = 0;
  for (int kb = 0; kb < K; kb += 32) {
    if (kb + 32 < K) load_tiles(kb + 32, cur ^ 1);   // prefetch next tile
    v16bf af[4], bfv[2];
#pragma unroll
    for (int i = 0; i < 4; ++i) af[i] = frag_a(lA[cur], mW + i * 16, lane);
#pragma unroll
    for (int j = 0; j < 2; ++j) bfv[j] = frag_b(lB[cur], nW + j * 16, lane);
#pragma unroll
    for (int i = 0; i < 4; ++i)
#pragma unroll
      for (int j = 0; j < 2; ++j)
        acc[i][j] = wmma_bf16(af[i], bfv[j], acc[i][j]);
    __syncthreads();
    cur ^= 1;
  }

  const int hf = lane >> 4, nc = lane & 15;
#pragma unroll
  for (int i = 0; i < 4; ++i)
#pragma unroll
    for (int j = 0; j < 2; ++j) {
      int gm0 = mBase + mW + i * 16 + hf * 8;
      int gn = nBase + nW + j * 16 + nc;
      float bv = HAS_BIAS ? bias[gn] : 0.0f;
#pragma unroll
      for (int r = 0; r < 8; ++r) {
        float v = acc[i][j][r] + bv;
        if (ACT == 1) v = fmaxf(v, 0.0f);
        if (ACT == 2) v *= scale;
        C[(size_t)(gm0 + r) * Nn + gn] = v;
      }
    }
}

// ---------------------------------------------------------------------------
// kv[b,h,f,d] = sum_s Kf[b,s,h,f] * V[b,s,h,d]   (A accessed transposed)
// Per (b,h): GEMM M=256(f) x N=64(d), K=1024(s). Grid (2, B*H), tile 128x64.
// ---------------------------------------------------------------------------
__global__ __launch_bounds__(256) void kv_wmma(
    const unsigned short* __restrict__ Kf, const float* __restrict__ V,
    float* __restrict__ kv)
{
  __shared__ __align__(16) unsigned short lA[2][128 * LDS_STRIDE]; // [f][s]
  __shared__ __align__(16) unsigned short lB[2][64 * LDS_STRIDE];  // [d][s]
  const int tid = threadIdx.x, lane = tid & 31, wave = tid >> 5;
  const int fBase = blockIdx.x * 128;
  const int bh = blockIdx.y;
  const int b = bh >> 3, hd = bh & 7;
  const int mW = (wave >> 1) * 32;
  const int nW = (wave & 1) * 32;
  v8f acc[2][2] = {};

  auto load_tiles = [&](int sb, int buf) {
    // Kf tile: (s 32) x (f 128) bf16, transpose into [f][s]
#pragma unroll
    for (int p = 0; p < 4; ++p) {
      int idx = p * 256 + tid;
      int sl = idx >> 5;
      int f4 = (idx & 31) * 4;
      size_t row = (size_t)((b * Ss + sb + sl) * Hh + hd);
      uint2 v = *(const uint2*)(Kf + row * Ff + fBase + f4);
      lA[buf][(size_t)(f4 + 0) * LDS_STRIDE + sl] = (unsigned short)(v.x & 0xffffu);
      lA[buf][(size_t)(f4 + 1) * LDS_STRIDE + sl] = (unsigned short)(v.x >> 16);
      lA[buf][(size_t)(f4 + 2) * LDS_STRIDE + sl] = (unsigned short)(v.y & 0xffffu);
      lA[buf][(size_t)(f4 + 3) * LDS_STRIDE + sl] = (unsigned short)(v.y >> 16);
    }
    // V tile: (s 32) x (d 64) f32 -> bf16, transpose into [d][s]
#pragma unroll
    for (int p = 0; p < 2; ++p) {
      int idx = p * 256 + tid;
      int sl = idx >> 4;
      int d4 = (idx & 15) * 4;
      float4 v = *(const float4*)(V + (size_t)(b * Ss + sb + sl) * DMm + hd * Dd + d4);
      lB[buf][(size_t)(d4 + 0) * LDS_STRIDE + sl] = f2bf(v.x);
      lB[buf][(size_t)(d4 + 1) * LDS_STRIDE + sl] = f2bf(v.y);
      lB[buf][(size_t)(d4 + 2) * LDS_STRIDE + sl] = f2bf(v.z);
      lB[buf][(size_t)(d4 + 3) * LDS_STRIDE + sl] = f2bf(v.w);
    }
  };

  load_tiles(0, 0);
  __syncthreads();
  int cur = 0;
  for (int sb = 0; sb < Ss; sb += 32) {
    if (sb + 32 < Ss) load_tiles(sb + 32, cur ^ 1);
    v16bf af[2], bfv[2];
    af[0] = frag_a(lA[cur], mW, lane);
    af[1] = frag_a(lA[cur], mW + 16, lane);
    bfv[0] = frag_b(lB[cur], nW, lane);
    bfv[1] = frag_b(lB[cur], nW + 16, lane);
#pragma unroll
    for (int i = 0; i < 2; ++i)
#pragma unroll
      for (int j = 0; j < 2; ++j)
        acc[i][j] = wmma_bf16(af[i], bfv[j], acc[i][j]);
    __syncthreads();
    cur ^= 1;
  }

  const int hf = lane >> 4, nc = lane & 15;
#pragma unroll
  for (int i = 0; i < 2; ++i)
#pragma unroll
    for (int j = 0; j < 2; ++j)
#pragma unroll
      for (int r = 0; r < 8; ++r) {
        int f = fBase + mW + i * 16 + hf * 8 + r;
        int d = nW + j * 16 + nc;
        kv[((size_t)bh * Ff + f) * Dd + d] = acc[i][j][r];
      }
}

// ---------------------------------------------------------------------------
// out[b,s,h,d] = z[b,s,h] * sum_f Qf[b,s,h,f] * kv[b,h,f,d]
// Per (b,h): GEMM M=1024(s) x N=64(d), K=256(f). Grid (8, B*H), tile 128x64.
// ---------------------------------------------------------------------------
__global__ __launch_bounds__(256) void qkv_wmma(
    const unsigned short* __restrict__ Qf, const float* __restrict__ kv,
    const float* __restrict__ z, float* __restrict__ out)
{
  __shared__ __align__(16) unsigned short lA[2][128 * LDS_STRIDE]; // [s][f]
  __shared__ __align__(16) unsigned short lB[2][64 * LDS_STRIDE];  // [d][f]
  const int tid = threadIdx.x, lane = tid & 31, wave = tid >> 5;
  const int sBase = blockIdx.x * 128;
  const int bh = blockIdx.y;
  const int b = bh >> 3, hd = bh & 7;
  const int mW = (wave >> 1) * 32;
  const int nW = (wave & 1) * 32;
  v8f acc[2][2] = {};

  auto load_tiles = [&](int kb, int buf) {
    // Qf tile: (s 128) x (f 32) bf16, straight copy into [s][f]
#pragma unroll
    for (int p = 0; p < 4; ++p) {
      int idx = p * 256 + tid;
      int sl = idx >> 3;
      int k4 = (idx & 7) * 4;
      size_t row = (size_t)((b * Ss + sBase + sl) * Hh + hd);
      uint2 v = *(const uint2*)(Qf + row * Ff + kb + k4);
      *(uint2*)(&lA[buf][(size_t)sl * LDS_STRIDE + k4]) = v;
    }
    // kv tile: (f 32) x (d 64) f32 -> bf16, transpose into [d][f]
#pragma unroll
    for (int p = 0; p < 2; ++p) {
      int idx = p * 256 + tid;
      int kl = idx >> 4;
      int d4 = (idx & 15) * 4;
      float4 v = *(const float4*)(kv + ((size_t)bh * Ff + kb + kl) * Dd + d4);
      lB[buf][(size_t)(d4 + 0) * LDS_STRIDE + kl] = f2bf(v.x);
      lB[buf][(size_t)(d4 + 1) * LDS_STRIDE + kl] = f2bf(v.y);
      lB[buf][(size_t)(d4 + 2) * LDS_STRIDE + kl] = f2bf(v.z);
      lB[buf][(size_t)(d4 + 3) * LDS_STRIDE + kl] = f2bf(v.w);
    }
  };

  load_tiles(0, 0);
  __syncthreads();
  int cur = 0;
  for (int kb = 0; kb < Ff; kb += 32) {
    if (kb + 32 < Ff) load_tiles(kb + 32, cur ^ 1);
    v16bf af[2], bfv[2];
    af[0] = frag_a(lA[cur], mW, lane);
    af[1] = frag_a(lA[cur], mW + 16, lane);
    bfv[0] = frag_b(lB[cur], nW, lane);
    bfv[1] = frag_b(lB[cur], nW + 16, lane);
#pragma unroll
    for (int i = 0; i < 2; ++i)
#pragma unroll
      for (int j = 0; j < 2; ++j)
        acc[i][j] = wmma_bf16(af[i], bfv[j], acc[i][j]);
    __syncthreads();
    cur ^= 1;
  }

  const int hf = lane >> 4, nc = lane & 15;
#pragma unroll
  for (int i = 0; i < 2; ++i)
#pragma unroll
    for (int j = 0; j < 2; ++j)
#pragma unroll
      for (int r = 0; r < 8; ++r) {
        int sg = sBase + mW + i * 16 + hf * 8 + r;
        size_t tok = (size_t)b * Ss + sg;
        float zv = z[tok * Hh + hd];
        int d = nW + j * 16 + nc;
        out[tok * DMm + hd * Dd + d] = acc[i][j][r] * zv;
      }
}

// ---------------------------------------------------------------------------
// Elementwise / small kernels
// ---------------------------------------------------------------------------

// Encoder first MLP: out[N,256] = relu(concat(x,pos) @ enc_w1 + enc_b1), K=6
__global__ __launch_bounds__(256) void enc1_kernel(
    const float* __restrict__ x, const float* __restrict__ pos,
    const float* __restrict__ w1, const float* __restrict__ b1,
    float* __restrict__ out)
{
  size_t idx = (size_t)blockIdx.x * 256 + threadIdx.x;
  int row = (int)(idx >> 8);
  int c = (int)(idx & 255);
  float a = b1[c];
#pragma unroll
  for (int j = 0; j < 3; ++j) {
    a += x[(size_t)row * 3 + j] * w1[j * 256 + c];
    a += pos[(size_t)row * 3 + j] * w1[(3 + j) * 256 + c];
  }
  out[idx] = fmaxf(a, 0.0f);
}

// LayerNorm over 512 cols with optional residual: out = LN(in (+ res)) * g + b
template <bool HAS_RES>
__global__ __launch_bounds__(256) void ln_kernel(
    const float* __restrict__ in, const float* __restrict__ res,
    const float* __restrict__ g, const float* __restrict__ bta,
    float* __restrict__ out)
{
  __shared__ float red[256];
  const int row = blockIdx.x, t = threadIdx.x;
  const size_t base = (size_t)row * DMm;
  float v0 = in[base + 2 * t], v1 = in[base + 2 * t + 1];
  if (HAS_RES) { v0 += res[base + 2 * t]; v1 += res[base + 2 * t + 1]; }
  red[t] = v0 + v1;
  __syncthreads();
  for (int o = 128; o > 0; o >>= 1) { if (t < o) red[t] += red[t + o]; __syncthreads(); }
  const float mean = red[0] * (1.0f / DMm);
  __syncthreads();
  float d0 = v0 - mean, d1 = v1 - mean;
  red[t] = d0 * d0 + d1 * d1;
  __syncthreads();
  for (int o = 128; o > 0; o >>= 1) { if (t < o) red[t] += red[t + o]; __syncthreads(); }
  const float rstd = rsqrtf(red[0] * (1.0f / DMm) + 1e-5f);
  out[base + 2 * t]     = d0 * rstd * g[2 * t]     + bta[2 * t];
  out[base + 2 * t + 1] = d1 * rstd * g[2 * t + 1] + bta[2 * t + 1];
}

// FAVOR+ feature map: for row r (= token*H + h), X[r,0:64] pre-scaled by D^-0.25,
// U[r,0:128] = X @ omega. Writes bf16 out[r, 0:256] = exp(+-u - h)/sqrt(F).
__global__ __launch_bounds__(128) void favor_kernel(
    const float* __restrict__ U, const float* __restrict__ X,
    unsigned short* __restrict__ out)
{
  __shared__ float red[64];
  __shared__ float hsh;
  const int r = blockIdx.x, t = threadIdx.x;
  if (t < 64) { float q = X[(size_t)r * 64 + t]; red[t] = q * q; }
  __syncthreads();
  if (t < 32) red[t] += red[t + 32];
  __syncthreads();
  if (t < 16) red[t] += red[t + 16];
  __syncthreads();
  if (t < 8) red[t] += red[t + 8];
  __syncthreads();
  if (t < 4) red[t] += red[t + 4];
  __syncthreads();
  if (t < 2) red[t] += red[t + 2];
  __syncthreads();
  if (t == 0) hsh = 0.5f * (red[0] + red[1]);
  __syncthreads();
  const float hv = hsh;
  const float u = U[(size_t)r * 128 + t];
  out[(size_t)r * Ff + t]       = f2bf(__expf(u - hv) * 0.0625f);   // 1/sqrt(256)
  out[(size_t)r * Ff + 128 + t] = f2bf(__expf(-u - hv) * 0.0625f);
}

// Ksum[b,h,f] = sum_s Kf[b,s,h,f]
__global__ __launch_bounds__(256) void ksum_kernel(
    const unsigned short* __restrict__ Kf, float* __restrict__ Ksum)
{
  const int bh = blockIdx.x, f = threadIdx.x;
  const int b = bh >> 3, hd = bh & 7;
  float s = 0.0f;
  for (int si = 0; si < Ss; ++si)
    s += bf2f(Kf[((size_t)((b * Ss + si) * Hh + hd)) * Ff + f]);
  Ksum[(size_t)bh * Ff + f] = s;
}

// z[r] = 1 / (Qf[r,:] . Ksum[b,h,:] + eps)
__global__ __launch_bounds__(256) void z_kernel(
    const unsigned short* __restrict__ Qf, const float* __restrict__ Ksum,
    float* __restrict__ z)
{
  const size_t r = (size_t)blockIdx.x * 256 + threadIdx.x;
  const int hd = (int)(r & 7);
  const size_t tok = r >> 3;
  const int b = (int)(tok >> 10);
  const unsigned short* q = Qf + r * Ff;
  const float* ks = Ksum + (size_t)(b * Hh + hd) * Ff;
  float d = 0.0f;
  for (int f = 0; f < Ff; ++f) d += bf2f(q[f]) * ks[f];
  z[r] = 1.0f / (d + 1e-6f);
}

// pooled[b,c] = mean_s h[b,s,c]
__global__ __launch_bounds__(256) void pool_kernel(
    const float* __restrict__ h, float* __restrict__ pooled)
{
  const int idx = blockIdx.x * 256 + threadIdx.x;   // 8192
  const int b = idx >> 9, c = idx & 511;
  float s = 0.0f;
  for (int si = 0; si < Ss; ++si)
    s += h[((size_t)(b * Ss + si)) * DMm + c];
  pooled[idx] = s * (1.0f / Ss);
}

// out[b,j] = pooled[b,:] @ cls_w[:,j] + cls_b[j]
__global__ __launch_bounds__(256) void cls_kernel(
    const float* __restrict__ pooled, const float* __restrict__ w,
    const float* __restrict__ bias, float* __restrict__ out)
{
  const int t = threadIdx.x;
  const int b = t >> 4, j = t & 15;
  float a = bias[j];
  for (int i = 0; i < DMm; ++i) a += pooled[b * DMm + i] * w[i * 16 + j];
  out[t] = a;
}

// ---------------------------------------------------------------------------
extern "C" void kernel_launch(void* const* d_in, const int* in_sizes, int n_in,
                              void* d_out, int out_size, void* d_ws, size_t ws_size,
                              hipStream_t stream)
{
  (void)in_sizes; (void)n_in; (void)out_size; (void)ws_size;
  const float* x      = (const float*)d_in[0];
  const float* pos    = (const float*)d_in[1];
  const float* enc_w1 = (const float*)d_in[2];
  const float* enc_b1 = (const float*)d_in[3];
  const float* enc_w2 = (const float*)d_in[4];
  const float* enc_b2 = (const float*)d_in[5];
  const float* enc_g  = (const float*)d_in[6];
  const float* enc_bb = (const float*)d_in[7];
  const float* wq  = (const float*)d_in[8];
  const float* bq  = (const float*)d_in[9];
  const float* wk  = (const float*)d_in[10];
  const float* bk  = (const float*)d_in[11];
  const float* wv  = (const float*)d_in[12];
  const float* bv  = (const float*)d_in[13];
  const float* wo  = (const float*)d_in[14];
  const float* bo  = (const float*)d_in[15];
  const float* ln1_g = (const float*)d_in[16];
  const float* ln1_b = (const float*)d_in[17];
  const float* w1  = (const float*)d_in[18];
  const float* b1  = (const float*)d_in[19];
  const float* w2  = (const float*)d_in[20];
  const float* b2  = (const float*)d_in[21];
  const float* ln2_g = (const float*)d_in[22];
  const float* ln2_b = (const float*)d_in[23];
  const float* omega = (const float*)d_in[24];
  const float* cls_w = (const float*)d_in[25];
  const float* cls_b = (const float*)d_in[26];

  // Workspace layout (bytes)
  char* wp = (char*)d_ws;
  size_t off = 0;
  float* h    = (float*)(wp + off); off += (size_t)Nt * DMm * 4;       // 33.5 MB
  float* qp   = (float*)(wp + off); off += (size_t)Nt * DMm * 4;
  float* kp   = (float*)(wp + off); off += (size_t)Nt * DMm * 4;
  float* vp   = (float*)(wp + off); off += (size_t)Nt * DMm * 4;
  float* attn = (float*)(wp + off); off += (size_t)Nt * DMm * 4;
  float* t1   = (float*)(wp + off); off += (size_t)Nt * 2 * DMm * 4;   // 67 MB (FFN mid / U / enc mid)
  unsigned short* Qf = (unsigned short*)(wp + off); off += (size_t)Nt * Hh * Ff * 2; // bf16
  unsigned short* Kf = (unsigned short*)(wp + off); off += (size_t)Nt * Hh * Ff * 2;
  float* kvb  = (float*)(wp + off); off += (size_t)Bb * Hh * Ff * Dd * 4;
  float* Ksum = (float*)(wp + off); off += (size_t)Bb * Hh * Ff * 4;
  float* zb   = (float*)(wp + off); off += (size_t)Nt * Hh * 4;
  float* pooled = (float*)(wp + off); off += (size_t)Bb * DMm * 4;

  const float SC = 0.35355339059327373f; // D^-0.25 = 64^-0.25

  // Encoder
  enc1_kernel<<<Nt, 256, 0, stream>>>(x, pos, enc_w1, enc_b1, t1);
  gemm_wmma<0, true><<<dim3(DMm / 128, Nt / 128), 256, 0, stream>>>(t1, enc_w2, enc_b2, attn, Nt, DMm, 256, 1.0f);
  ln_kernel<false><<<Nt, 256, 0, stream>>>(attn, nullptr, enc_g, enc_bb, h);

  for (int l = 0; l < Ll; ++l) {
    const float* wql = wq + (size_t)l * DMm * DMm;
    const float* wkl = wk + (size_t)l * DMm * DMm;
    const float* wvl = wv + (size_t)l * DMm * DMm;
    const float* wol = wo + (size_t)l * DMm * DMm;
    const float* bql = bq + (size_t)l * DMm;
    const float* bkl = bk + (size_t)l * DMm;
    const float* bvl = bv + (size_t)l * DMm;
    const float* bol = bo + (size_t)l * DMm;
    const float* w1l = w1 + (size_t)l * DMm * 2 * DMm;
    const float* b1l = b1 + (size_t)l * 2 * DMm;
    const float* w2l = w2 + (size_t)l * 2 * DMm * DMm;
    const float* b2l = b2 + (size_t)l * DMm;
    const float* omgl = omega + (size_t)l * Dd * (Ff / 2);

    // Projections (scale D^-0.25 folded into q/k epilogue)
    gemm_wmma<2, true><<<dim3(4, Nt / 128), 256, 0, stream>>>(h, wql, bql, qp, Nt, DMm, DMm, SC);
    gemm_wmma<2, true><<<dim3(4, Nt / 128), 256, 0, stream>>>(h, wkl, bkl, kp, Nt, DMm, DMm, SC);
    gemm_wmma<0, true><<<dim3(4, Nt / 128), 256, 0, stream>>>(h, wvl, bvl, vp, Nt, DMm, DMm, 1.0f);

    // FAVOR features: U = X @ omega  (M = Nt*H = 131072, K = 64, N = 128)
    gemm_wmma<0, false><<<dim3(1, (Nt * Hh) / 128), 256, 0, stream>>>(qp, omgl, nullptr, t1, Nt * Hh, 128, Dd, 1.0f);
    favor_kernel<<<Nt * Hh, 128, 0, stream>>>(t1, qp, Qf);
    gemm_wmma<0, false><<<dim3(1, (Nt * Hh) / 128), 256, 0, stream>>>(kp, omgl, nullptr, t1, Nt * Hh, 128, Dd, 1.0f);
    favor_kernel<<<Nt * Hh, 128, 0, stream>>>(t1, kp, Kf);

    // Linear attention
    ksum_kernel<<<Bb * Hh, 256, 0, stream>>>(Kf, Ksum);
    kv_wmma<<<dim3(2, Bb * Hh), 256, 0, stream>>>(Kf, vp, kvb);
    z_kernel<<<(Nt * Hh) / 256, 256, 0, stream>>>(Qf, Ksum, zb);
    qkv_wmma<<<dim3(8, Bb * Hh), 256, 0, stream>>>(Qf, kvb, zb, attn);

    // Output projection + post-norm
    gemm_wmma<0, true><<<dim3(4, Nt / 128), 256, 0, stream>>>(attn, wol, bol, t1, Nt, DMm, DMm, 1.0f);
    ln_kernel<true><<<Nt, 256, 0, stream>>>(t1, h, ln1_g + (size_t)l * DMm, ln1_b + (size_t)l * DMm, h);

    // FFN + post-norm
    gemm_wmma<1, true><<<dim3(8, Nt / 128), 256, 0, stream>>>(h, w1l, b1l, t1, Nt, 2 * DMm, DMm, 1.0f);
    gemm_wmma<0, true><<<dim3(4, Nt / 128), 256, 0, stream>>>(t1, w2l, b2l, attn, Nt, DMm, 2 * DMm, 1.0f);
    ln_kernel<true><<<Nt, 256, 0, stream>>>(attn, h, ln2_g + (size_t)l * DMm, ln2_b + (size_t)l * DMm, h);
  }

  // Pool + classify
  pool_kernel<<<(Bb * DMm) / 256, 256, 0, stream>>>(h, pooled);
  cls_kernel<<<1, 256, 0, stream>>>(pooled, cls_w, cls_b, (float*)d_out);
}
